// AttentionLayer_88545045774526
// MI455X (gfx1250) — matrix operand order repaired and checked by reading, not compiled, not run on hardware
//
#include <hip/hip_runtime.h>
#include <hip/hip_bf16.h>

// ---------------------------------------------------------------------------
// Types
// ---------------------------------------------------------------------------
typedef __attribute__((ext_vector_type(16))) __bf16 bf16x16;
typedef __attribute__((ext_vector_type(8)))  float  f32x8;

union Frag {
    uint4   u4[2];   // 32 bytes
    bf16x16 v;       // 16 x bf16
};

__device__ __forceinline__ uint4 ld16B(const __bf16* p) {
    return *reinterpret_cast<const uint4*>(p);
}

#define WMMA_BF16(A, B, C) \
    __builtin_amdgcn_wmma_f32_16x16x32_bf16(false, (A), false, (B), (short)0, (C), false, false)

// Problem constants
#define BB   2
#define NN   2048
#define DD   1024
#define HH   16
#define HD   64
#define ROWS (BB * NN)          // 4096 flattened token rows
#define QKVC (3 * DD)           // 3072

// ---------------------------------------------------------------------------
// fp32 -> bf16 convert (vectorized: float4 in, 4 x bf16 out)
// ---------------------------------------------------------------------------
__global__ void cvt_f32_bf16_kernel(const float* __restrict__ src,
                                    __bf16* __restrict__ dst, int n4) {
    int i = blockIdx.x * blockDim.x + threadIdx.x;
    int stride = gridDim.x * blockDim.x;
    for (; i < n4; i += stride) {
        float4 f = reinterpret_cast<const float4*>(src)[i];
        __bf16 o[4] = {(__bf16)f.x, (__bf16)f.y, (__bf16)f.z, (__bf16)f.w};
        *reinterpret_cast<uint2*>(dst + (size_t)i * 4) = *reinterpret_cast<uint2*>(o);
    }
}

// ---------------------------------------------------------------------------
// Tiled bf16 GEMM:  C[M,N] = A[M,K] @ B[K,N] + bias[N]
// Block: 256 threads (8 waves), tile 128(M) x 128(N), K-step 32.
// Wave grid 4(M) x 2(N); each wave computes 32x64 = 2x4 WMMA tiles.
// Double-buffered LDS: A staged [128][32] row-major; B staged transposed
// [128 n][32 k] so both A and B fragments are contiguous 16B LDS loads per
// the 16-bit WMMA VGPR layouts. One barrier per K-step; next tile's global
// loads overlap current tile's WMMAs; prefetch two tiles ahead.
// ---------------------------------------------------------------------------
struct Stage {
    uint4 a[2];
    uint4 b[2];
};

template <bool OUT_F32>
__global__ __launch_bounds__(256)
void gemm_bf16_kernel(const __bf16* __restrict__ A, const __bf16* __restrict__ B,
                      const float* __restrict__ bias,
                      float* __restrict__ Cf, __bf16* __restrict__ Cb,
                      int M, int N, int K) {
    __shared__ __align__(16) __bf16 sA[2][128 * 32];
    __shared__ __align__(16) __bf16 sBT[2][128 * 32];

    const int tid   = threadIdx.x;
    const int lane  = tid & 31;
    const int hf    = (lane >> 4) & 1;     // which 16-lane half of the wave
    const int lrow  = lane & 15;
    const int wave  = tid >> 5;
    const int waveM = wave >> 1;           // 0..3
    const int waveN = wave & 1;            // 0..1
    const int mBase = blockIdx.y * 128;
    const int nBase = blockIdx.x * 128;

    // Per-thread staging coordinates (2 x uint4 each for A and B)
    int aRow[2], aCol[2], bKr[2], bNc[2], aIdx[2];
#pragma unroll
    for (int r = 0; r < 2; ++r) {
        int idx = (tid + r * 256) * 8;     // bf16 element index within tile
        aIdx[r] = idx;
        aRow[r] = idx >> 5;  aCol[r] = idx & 31;
        bKr[r]  = idx >> 7;  bNc[r]  = idx & 127;
    }

    auto loadStage = [&](int kb) {
        Stage s;
#pragma unroll
        for (int r = 0; r < 2; ++r) {
            s.a[r] = ld16B(&A[(size_t)(mBase + aRow[r]) * K + kb + aCol[r]]);
            s.b[r] = ld16B(&B[(size_t)(kb + bKr[r]) * N + nBase + bNc[r]]);
        }
        return s;
    };
    auto storeStage = [&](const Stage& s, int buf) {
#pragma unroll
        for (int r = 0; r < 2; ++r)
            *reinterpret_cast<uint4*>(&sA[buf][aIdx[r]]) = s.a[r];
#pragma unroll
        for (int r = 0; r < 2; ++r) {
            const __bf16* e = reinterpret_cast<const __bf16*>(&s.b[r]);
#pragma unroll
            for (int j = 0; j < 8; ++j) sBT[buf][(bNc[r] + j) * 32 + bKr[r]] = e[j];
        }
    };

    f32x8 acc[2][4];
#pragma unroll
    for (int mt = 0; mt < 2; ++mt)
#pragma unroll
        for (int nt = 0; nt < 4; ++nt) acc[mt][nt] = 0.0f;

    // Prologue: stage first K tile
    {
        Stage s0 = loadStage(0);
        storeStage(s0, 0);
    }
    __syncthreads();

    int pb = 0;
    for (int kb = 0; kb < K; kb += 32, pb ^= 1) {
        const bool more = (kb + 32) < K;
        Stage stn;
        if (more) {
            stn = loadStage(kb + 32);
            if (kb + 64 < K) {      // prefetch two tiles ahead -> global_prefetch_b8
                __builtin_prefetch(&A[(size_t)(mBase + aRow[0]) * K + kb + 64 + aCol[0]], 0, 0);
                __builtin_prefetch(&B[(size_t)(kb + 64 + bKr[0]) * N + nBase + bNc[0]], 0, 0);
            }
        }

        // ---- fragment loads from current buffer ----
        Frag af[2], bfr[4];
#pragma unroll
        for (int mt = 0; mt < 2; ++mt) {
            const __bf16* p = &sA[pb][(waveM * 32 + mt * 16 + lrow) * 32];
            af[mt].u4[0] = ld16B(p + 8 * hf);        // K = 8h .. 8h+7
            af[mt].u4[1] = ld16B(p + 16 + 8 * hf);   // K = 16+8h .. 16+8h+7
        }
#pragma unroll
        for (int nt = 0; nt < 4; ++nt) {
            const __bf16* p = &sBT[pb][(waveN * 64 + nt * 16 + lrow) * 32];
            bfr[nt].u4[0] = ld16B(p + 8 * hf);       // K = 8h .. 8h+7
            bfr[nt].u4[1] = ld16B(p + 16 + 8 * hf);  // K = 16+8h .. 16+8h+7
        }
        // ---- 8 WMMAs ----
#pragma unroll
        for (int mt = 0; mt < 2; ++mt)
#pragma unroll
            for (int nt = 0; nt < 4; ++nt)
                acc[mt][nt] = WMMA_BF16(af[mt].v, bfr[nt].v, acc[mt][nt]);

        if (more) storeStage(stn, pb ^ 1);
        __syncthreads();
    }

    // ---- epilogue: C layout lane=n, VGPR v -> m = v + 8*half ----
#pragma unroll
    for (int mt = 0; mt < 2; ++mt) {
#pragma unroll
        for (int nt = 0; nt < 4; ++nt) {
            int n = nBase + waveN * 64 + nt * 16 + lrow;
            float bv = bias ? bias[n] : 0.0f;
#pragma unroll
            for (int v = 0; v < 8; ++v) {
                int m = mBase + waveM * 32 + mt * 16 + v + 8 * hf;
                float val = acc[mt][nt][v] + bv;
                if (OUT_F32) Cf[(size_t)m * N + n] = val;
                else         Cb[(size_t)m * N + n] = (__bf16)val;
            }
        }
    }
}

// ---------------------------------------------------------------------------
// Flash attention (causal), bf16 QKV with fp32 softmax/accum.
// Grid: (N/64, H, B); block: 128 threads = 4 waves; each wave owns 16 q rows.
// Computes S^T = K @ Q^T so the post-softmax P converts to the WMMA A-layout
// purely in-lane. V is staged transposed in (double-buffered) LDS for the
// P@V B fragments. K fragments for tile t+1 are prefetched into registers
// during tile t's compute; one barrier per kv tile.
// ---------------------------------------------------------------------------
struct KFrags {
    Frag f[2][2];   // [kv m-tile][K chunk]
};

__global__ __launch_bounds__(128)
void attn_kernel(const __bf16* __restrict__ qkv, __bf16* __restrict__ out) {
    __shared__ __align__(16) __bf16 sVT[2][HD * 32];  // [buf][hd][kv] transposed V

    const int tid  = threadIdx.x;
    const int lane = tid & 31;
    const int hf   = (lane >> 4) & 1;
    const int lrow = lane & 15;
    const int wave = tid >> 5;

    const int b       = blockIdx.z;
    const int h       = blockIdx.y;
    const int qTile   = blockIdx.x * 64;
    const int rowBase = b * NN;
    const int qPos    = qTile + wave * 16 + lrow;   // this lane's q column of S^T
    const int qRow    = rowBase + qPos;

    // Q^T B-fragments, K(=HD) split into two 32-chunks (loaded once)
    Frag qb[2];
#pragma unroll
    for (int kc = 0; kc < 2; ++kc) {
        const __bf16* qp = qkv + (size_t)qRow * QKVC + h * HD + kc * 32 + 8 * hf;
        qb[kc].u4[0] = ld16B(qp);
        qb[kc].u4[1] = ld16B(qp + 16);
    }

    // Cooperative V staging: thread -> (kv row, 16 hd columns), transposed
    const int vKvr = tid >> 2;          // 0..31
    const int vHd0 = (tid & 3) * 16;
    auto stageV = [&](int kvBase, int buf) {
        const __bf16* vp =
            qkv + (size_t)(rowBase + kvBase + vKvr) * QKVC + 2 * DD + h * HD + vHd0;
        uint4 d0 = ld16B(vp), d1 = ld16B(vp + 8);
        const __bf16* e0 = reinterpret_cast<const __bf16*>(&d0);
        const __bf16* e1 = reinterpret_cast<const __bf16*>(&d1);
#pragma unroll
        for (int j = 0; j < 8; ++j) {
            sVT[buf][(vHd0 + j) * 32 + vKvr]     = e0[j];
            sVT[buf][(vHd0 + 8 + j) * 32 + vKvr] = e1[j];
        }
    };
    // K A-fragments for one 32-row kv tile (register prefetchable)
    auto loadK = [&](int kvBase) {
        KFrags kf;
#pragma unroll
        for (int mt = 0; mt < 2; ++mt) {
            const __bf16* kp = qkv +
                (size_t)(rowBase + kvBase + mt * 16 + lrow) * QKVC + DD + h * HD;
#pragma unroll
            for (int kc = 0; kc < 2; ++kc) {
                kf.f[mt][kc].u4[0] = ld16B(kp + kc * 32 + 8 * hf);
                kf.f[mt][kc].u4[1] = ld16B(kp + kc * 32 + 16 + 8 * hf);
            }
        }
        return kf;
    };

    f32x8 o[4];
#pragma unroll
    for (int t4 = 0; t4 < 4; ++t4) o[t4] = 0.0f;
    float r_max = -INFINITY, r_sum = 0.0f;
    const float scale = 0.125f;                        // 1/sqrt(64)

    const int nKv = qTile / 32 + 2;                    // causal: kv in [0, qTile+64)

    // Prologue: stage tile 0
    stageV(0, 0);
    KFrags kf = loadK(0);
    __syncthreads();

    for (int t = 0; t < nKv; ++t) {
        const int kvBase = t * 32;
        const int buf = t & 1;
        const bool more = (t + 1) < nKv;

        KFrags kfn = kf;
        if (more) {                 // overlap next tile's loads with compute
            stageV((t + 1) * 32, buf ^ 1);
            kfn = loadK((t + 1) * 32);
        }

        // ---- S^T = K @ Q^T : 2 kv m-tiles x 2 K-chunks = 4 WMMAs ----
        f32x8 c[2];
        c[0] = 0.0f; c[1] = 0.0f;
#pragma unroll
        for (int mt = 0; mt < 2; ++mt)
#pragma unroll
            for (int kc = 0; kc < 2; ++kc)
                c[mt] = WMMA_BF16(kf.f[mt][kc].v, qb[kc].v, c[mt]);

        // ---- online softmax (rows = q, lane-indexed; kv across VGPRs/halves)
        float p0[8], p1[8];
        float tmax = -INFINITY;
#pragma unroll
        for (int v = 0; v < 8; ++v) {
            int kv0 = kvBase + v + 8 * hf;
            int kv1 = kv0 + 16;
            float s0 = (kv0 > qPos) ? -INFINITY : c[0][v] * scale;
            float s1 = (kv1 > qPos) ? -INFINITY : c[1][v] * scale;
            p0[v] = s0; p1[v] = s1;
            tmax = fmaxf(tmax, fmaxf(s0, s1));
        }
        tmax = fmaxf(tmax, __shfl_xor(tmax, 16, 32));  // combine halves of row
        float nmax = fmaxf(r_max, tmax);
        float corr = __expf(r_max - nmax);
        r_max = nmax;

        Frag pa;                                        // P in A-layout (in-lane!)
        float psum = 0.0f;
#pragma unroll
        for (int v = 0; v < 8; ++v) {
            float e0 = __expf(p0[v] - nmax);
            float e1 = __expf(p1[v] - nmax);
            psum += e0 + e1;
            pa.v[v]     = (__bf16)e0;                  // elems 0..7  -> kv 8h..8h+7
            pa.v[v + 8] = (__bf16)e1;                  // elems 8..15 -> kv 16+8h..
        }
        psum += __shfl_xor(psum, 16, 32);
        r_sum = r_sum * corr + psum;

        // rescale O (O rows q = v + 8*half -> broadcast corr from lane q)
        float corrv[8];
#pragma unroll
        for (int v = 0; v < 8; ++v) corrv[v] = __shfl(corr, v + 8 * hf, 32);
#pragma unroll
        for (int t4 = 0; t4 < 4; ++t4)
#pragma unroll
            for (int v = 0; v < 8; ++v) o[t4][v] *= corrv[v];

        // ---- O += P @ V : 4 hd n-tiles ----
#pragma unroll
        for (int t4 = 0; t4 < 4; ++t4) {
            Frag vb;
            const __bf16* vtp = &sVT[buf][(t4 * 16 + lrow) * 32 + 8 * hf];
            vb.u4[0] = ld16B(vtp);
            vb.u4[1] = ld16B(vtp + 16);
            o[t4] = WMMA_BF16(pa.v, vb.v, o[t4]);
        }

        __syncthreads();
        kf = kfn;
    }

    // ---- finalize: O /= row_sum, store bf16 [4096, 1024] ----
    float inv = 1.0f / r_sum;
    float invv[8];
#pragma unroll
    for (int v = 0; v < 8; ++v) invv[v] = __shfl(inv, v + 8 * hf, 32);
#pragma unroll
    for (int t4 = 0; t4 < 4; ++t4) {
#pragma unroll
        for (int v = 0; v < 8; ++v) {
            int m   = qTile + wave * 16 + v + 8 * hf;
            int col = h * HD + t4 * 16 + lrow;
            out[(size_t)(rowBase + m) * DD + col] = (__bf16)(o[t4][v] * invv[v]);
        }
    }
}

// ---------------------------------------------------------------------------
// Launch
// ---------------------------------------------------------------------------
extern "C" void kernel_launch(void* const* d_in, const int* in_sizes, int n_in,
                              void* d_out, int out_size, void* d_ws, size_t ws_size,
                              hipStream_t stream) {
    (void)in_sizes; (void)n_in; (void)out_size; (void)ws_size;

    const float* X  = (const float*)d_in[0];   // [2,2048,1024]
    const float* Wa = (const float*)d_in[1];   // [1024,3072]
    const float* ba = (const float*)d_in[2];   // [3072]
    const float* Wp = (const float*)d_in[3];   // [1024,1024]
    const float* bp = (const float*)d_in[4];   // [1024]
    float* outF     = (float*)d_out;           // [2,2048,1024]

    char* ws = (char*)d_ws;
    __bf16* Xb  = (__bf16*)(ws);                              //  8 MB
    __bf16* Wab = (__bf16*)(ws + 8388608);                    //  6 MB
    __bf16* Wpb = (__bf16*)(ws + 14680064);                   //  2 MB
    __bf16* QKV = (__bf16*)(ws + 16777216);                   // 24 MB
    __bf16* AO  = (__bf16*)(ws + 41943040);                   //  8 MB

    // fp32 -> bf16 conversions (vectorized by 4)
    cvt_f32_bf16_kernel<<<1024, 256, 0, stream>>>(X,  Xb,  ROWS * DD / 4);
    cvt_f32_bf16_kernel<<<1024, 256, 0, stream>>>(Wa, Wab, DD * QKVC / 4);
    cvt_f32_bf16_kernel<<<512,  256, 0, stream>>>(Wp, Wpb, DD * DD / 4);

    // QKV projection: [4096,1024] @ [1024,3072] + b_attn -> bf16 QKV
    gemm_bf16_kernel<false><<<dim3(QKVC / 128, ROWS / 128), 256, 0, stream>>>(
        Xb, Wab, ba, nullptr, QKV, ROWS, QKVC, DD);

    // causal flash attention per (q-tile, head, batch)
    attn_kernel<<<dim3(NN / 64, HH, BB), 128, 0, stream>>>(QKV, AO);

    // output projection: [4096,1024] @ [1024,1024] + b_proj -> fp32 out
    gemm_bf16_kernel<true><<<dim3(DD / 128, ROWS / 128), 256, 0, stream>>>(
        AO, Wpb, bp, outF, nullptr, ROWS, DD, DD);
}
